// EdgeConvModel_75737453297751
// MI455X (gfx1250) — matrix operand order repaired and checked by reading, not compiled, not run on hardware
//
#include <hip/hip_runtime.h>
#include <hip/hip_bf16.h>
#include <math.h>

// ---------------------------------------------------------------------------
// EdgeConv x3 + maxpool + linear for MI455X (gfx1250, wave32, WMMA).
//
//  * concat(x_i, x_j-x_i)@wa  ==  P[dst] + Q[src], with per-NODE GEMMs
//      P = x @ (wa_top - wa_bot),  Q = x @ wa_bot        (16x fewer rows)
//  * relu(segment_max(msg)) with neginf->0  ==  atomicMax(int) of
//      max(msg,0) into a zero-initialized buffer (bit-order == value-order
//      for non-negative IEEE floats; idempotent => deterministic replay).
//  * per-edge second GEMM [16,128]@[128,128] via v_wmma_f32_16x16x32_f16.
//    A fragments built in registers (each lane owns exactly the 64 channels
//    of one edge its fragment needs); B fragments served from an LDS-pinned
//    copy of the weights so the random PQ gathers can't evict them and the
//    vmem path is dedicated to the bandwidth-critical gather/scatter.
// ---------------------------------------------------------------------------

typedef __attribute__((ext_vector_type(16))) _Float16 v16h;
typedef __attribute__((ext_vector_type(8)))  _Float16 v8h;
typedef __attribute__((ext_vector_type(8)))  float    v8f;

#define N_NODES 100000
#define N_EDGES 1600000LL
#define OUT_CH  128
#define WB_PAD  136   // 128 + 8 halves: column stride 68 dwords -> conflict-free quads

// ---------------- utility / prep kernels ----------------

__global__ void zero_f32(float* __restrict__ p, size_t n) {
  size_t i = (size_t)blockIdx.x * blockDim.x + threadIdx.x;
  if (i < n) p[i] = 0.0f;
}

// Build transposed, f16, K-padded node weight Wt[col][k], col in [0,256):
//   col <  128 : Wd = wa_top - wa_bot   (multiplies x_i==x[dst])
//   col >= 128 : Wq = wa_bot            (multiplies x_j==x[src])
__global__ void prep_wa(const float* __restrict__ wa, _Float16* __restrict__ Wt,
                        int C, int K) {
  int idx = blockIdx.x * blockDim.x + threadIdx.x;
  if (idx >= 256 * K) return;
  int col = idx / K, k = idx % K;
  float v = 0.0f;
  if (k < C) {
    int n = col & 127;
    float top = wa[k * OUT_CH + n];
    float bot = wa[(C + k) * OUT_CH + n];
    v = (col < 128) ? (top - bot) : bot;
  }
  Wt[idx] = (_Float16)v;
}

// Transposed f16 second-linear weight: Wbt[n][k] = wb[k][n]
__global__ void prep_wb(const float* __restrict__ wb, _Float16* __restrict__ Wbt) {
  int idx = blockIdx.x * blockDim.x + threadIdx.x;
  if (idx >= 128 * 128) return;
  int col = idx / 128, k = idx % 128;
  Wbt[idx] = (_Float16)wb[k * 128 + col];
}

// Input features f32 [N,24] -> f16 [N,32] zero-padded along K
__global__ void cvt_x_in(const float* __restrict__ x, _Float16* __restrict__ xh) {
  int idx = blockIdx.x * blockDim.x + threadIdx.x;
  if (idx >= N_NODES * 32) return;
  int node = idx >> 5, k = idx & 31;
  xh[idx] = (k < 24) ? (_Float16)x[node * 24 + k] : (_Float16)0.0f;
}

__global__ void cvt_f32_f16(const float* __restrict__ src, _Float16* __restrict__ dst,
                            size_t n) {
  size_t i = (size_t)blockIdx.x * blockDim.x + threadIdx.x;
  if (i < n) dst[i] = (_Float16)src[i];
}

// ---------------- node-side GEMM: Y[N,256](f16) = X[N,K](f16) @ Wt^T ----------------
// Wave tile: 16 rows x 256 cols. 4 waves / 128-thread block. A fragments are
// hoisted out of the N-tile loop (invariant across the 16 tiles).
template <int K>
__global__ __launch_bounds__(128)
void node_gemm(const _Float16* __restrict__ X, const _Float16* __restrict__ Wt,
               _Float16* __restrict__ Y, int nrows) {
  const int lane    = threadIdx.x & 31;
  const int wave    = threadIdx.x >> 5;
  const int rowBase = (blockIdx.x * 4 + wave) * 16;
  if (rowBase >= nrows) return;              // wave-uniform: EXEC stays all-1 for WMMA

  const int r16    = lane & 15;
  const int hiHalf = lane >> 4;              // 0: lanes 0-15, 1: lanes 16-31
  const int aOff   = hiHalf * 8;             // A-fragment K sub-offset per ISA layout
  const int bOff   = hiHalf * 8;             // B-fragment K sub-offset per ISA layout
  int aRow = rowBase + r16;
  if (aRow >= nrows) aRow = nrows - 1;       // duplicate row: benign (same data)
  const _Float16* xrow = X + (size_t)aRow * K;
  constexpr int nK = K >> 5;

  // A 16x32 fragments, one per K-step, invariant across all 16 N-tiles
  v16h afrag[nK];
#pragma unroll
  for (int ks = 0; ks < nK; ++ks) {
    const _Float16* ap = xrow + ks * 32 + aOff;
    v8h alo = *(const v8h*)(ap);             // k = ks*32 + aOff + 0..7
    v8h ahi = *(const v8h*)(ap + 16);        // k = ks*32 + 16 + aOff + 0..7
#pragma unroll
    for (int i = 0; i < 8; ++i) { afrag[ks][i] = alo[i]; afrag[ks][i + 8] = ahi[i]; }
  }

#pragma unroll 2
  for (int nt = 0; nt < 16; ++nt) {
    const int col = nt * 16 + r16;
    const _Float16* wcol = Wt + (size_t)col * K;
    v8f acc = {};
#pragma unroll
    for (int ks = 0; ks < nK; ++ks) {
      // B 32x16 fragment: k = ks*32+bOff+0..7 and ks*32+16+bOff+0..7 (weights pre-transposed)
      v8h blo = *(const v8h*)(wcol + ks * 32 + bOff);
      v8h bhi = *(const v8h*)(wcol + ks * 32 + bOff + 16);
      v16h b;
#pragma unroll
      for (int i = 0; i < 8; ++i) { b[i] = blo[i]; b[i + 8] = bhi[i]; }
      acc = __builtin_amdgcn_wmma_f32_16x16x32_f16(false, afrag[ks], false, b,
                                                   (short)0, acc, false, false);
    }
    // D layout: vgpr r -> row m = r + 8*hiHalf, col = nt*16 + (lane&15)
#pragma unroll
    for (int r = 0; r < 8; ++r) {
      int orow = rowBase + r + (hiHalf ? 8 : 0);
      if (orow < nrows) Y[(size_t)orow * 256 + col] = (_Float16)acc[r];
    }
  }
}

// ---------------- per-edge kernel: h=relu(P[dst]+Q[src]+ba); msg=h@Wb+bb ----------------
// Wave tile: 16 edges x 128 out channels; atomic-max scatter into outAgg[dst].
// Wbt is pinned in LDS so the random PQ gather stream cannot evict it.
__global__ __launch_bounds__(128)
void edge_conv(const _Float16* __restrict__ PQ,      // [N,256] f16: P | Q
               const long long* __restrict__ ei,     // [2,E] int64: src row0, dst row1
               const float* __restrict__ ba,         // [128]
               const _Float16* __restrict__ Wbt,     // [128,128] transposed f16
               const float* __restrict__ bb,         // [128]
               float* __restrict__ outAgg,           // [N,128] f32, pre-zeroed
               long long E) {
  __shared__ __align__(16) _Float16 ldsW[128 * WB_PAD];

  const int tid  = threadIdx.x;
  const int lane = tid & 31;
  const int wave = tid >> 5;

  __builtin_prefetch(Wbt, 0, 3);             // global_prefetch_b8

  // Cooperative stage of the 32KB weight tile into LDS (padded columns).
  // 2048 x 16B chunks, 16 per thread.
#pragma unroll 4
  for (int c = tid; c < 128 * 16; c += 128) {
    int col = c >> 4;
    int k0  = (c & 15) * 8;
    *(v8h*)(&ldsW[col * WB_PAD + k0]) = *(const v8h*)(Wbt + col * 128 + k0);
  }
  __syncthreads();                           // uniform: no wave exits before here

  long long tileBase = ((long long)blockIdx.x * 4 + wave) * 16;
  if (tileBase >= E) tileBase = 0;           // overflow waves redo tile 0 (idempotent)

  const int r16    = lane & 15;
  const int hiHalf = lane >> 4;
  const int aOff   = hiHalf * 8;
  const int bOff   = hiHalf * 8;

  // This lane's A row is edge (tileBase + r16); its fragments need exactly
  // the 64 channels {ks*32 + g*16 + aOff + 0..7 : ks<4, g<2} of h for that edge.
  long long ge = tileBase + r16;
  if (ge >= E) ge = E - 1;                   // duplicate edge: max is idempotent
  const long long srcN = ei[ge];
  const long long dstN = ei[E + ge];
  const _Float16* Prow = PQ + (size_t)dstN * 256;        // P block (cols 0..127)
  const _Float16* Qrow = PQ + (size_t)srcN * 256 + 128;  // Q block (cols 128..255)

  v16h afrag[4];
#pragma unroll
  for (int ks = 0; ks < 4; ++ks) {
#pragma unroll
    for (int g = 0; g < 2; ++g) {
      const int cbase = ks * 32 + g * 16 + aOff;
      v8h p = *(const v8h*)(Prow + cbase);
      v8h q = *(const v8h*)(Qrow + cbase);
#pragma unroll
      for (int i = 0; i < 8; ++i) {
        float h = (float)p[i] + (float)q[i] + ba[cbase + i];
        afrag[ks][g * 8 + i] = (_Float16)fmaxf(h, 0.0f);
      }
    }
  }

  // dst node ids of the 8 D-rows this lane will write (m = r + 8*hiHalf)
  int dstRows[8];
#pragma unroll
  for (int r = 0; r < 8; ++r) {
    long long geo = tileBase + r + (hiHalf ? 8 : 0);
    if (geo >= E) geo = E - 1;
    dstRows[r] = (int)ei[E + geo];
  }

#pragma unroll
  for (int nt = 0; nt < 8; ++nt) {
    const int col = nt * 16 + r16;
    const _Float16* wcol = &ldsW[col * WB_PAD];
    v8f acc = {};
#pragma unroll
    for (int ks = 0; ks < 4; ++ks) {
      // B fragment from LDS: k = bOff+0..7 and 16+bOff+0..7 as two 16B ds loads
      v8h blo = *(const v8h*)(wcol + ks * 32 + bOff);
      v8h bhi = *(const v8h*)(wcol + ks * 32 + bOff + 16);
      v16h b;
#pragma unroll
      for (int i = 0; i < 8; ++i) { b[i] = blo[i]; b[i + 8] = bhi[i]; }
      acc = __builtin_amdgcn_wmma_f32_16x16x32_f16(false, afrag[ks], false, b,
                                                   (short)0, acc, false, false);
    }
    const float bias = bb[col];
#pragma unroll
    for (int r = 0; r < 8; ++r) {
      float v = fmaxf(acc[r] + bias, 0.0f);  // relu folded into max-aggregation
      atomicMax((int*)(outAgg + (size_t)dstRows[r] * 128 + col),
                __float_as_int(v));          // non-negative: int order == float order
    }
  }
}

// ---------------- final: maxpool(384 -> 16 groups of 24) then @fw + fb ----------------
__global__ void final_head(const float* __restrict__ x2, const float* __restrict__ x4,
                           const float* __restrict__ x6, const float* __restrict__ fw,
                           const float* __restrict__ fb, float* __restrict__ out, int n) {
  int node = blockIdx.x * blockDim.x + threadIdx.x;
  if (node >= n) return;
  const float* a = x2 + (size_t)node * 128;
  const float* b = x4 + (size_t)node * 128;
  const float* c = x6 + (size_t)node * 128;
  float s = fb[0];
#pragma unroll
  for (int g = 0; g < 16; ++g) {
    float m = -3.402823466e38f;
#pragma unroll
    for (int j = 0; j < 24; ++j) {
      int ch = g * 24 + j;                   // channel in the 384-wide concat
      float v = (ch < 128) ? a[ch] : (ch < 256) ? b[ch - 128] : c[ch - 256];
      m = fmaxf(m, v);
    }
    s += m * fw[g];
  }
  out[node] = s;
}

// ---------------------------------------------------------------------------

extern "C" void kernel_launch(void* const* d_in, const int* in_sizes, int n_in,
                              void* d_out, int out_size, void* d_ws, size_t ws_size,
                              hipStream_t stream) {
  const float*     x   = (const float*)d_in[0];
  const long long* ei  = (const long long*)d_in[1];   // int64 edge_index [2,E]
  const float* w1a = (const float*)d_in[2];  const float* b1a = (const float*)d_in[3];
  const float* w1b = (const float*)d_in[4];  const float* b1b = (const float*)d_in[5];
  const float* w2a = (const float*)d_in[6];  const float* b2a = (const float*)d_in[7];
  const float* w2b = (const float*)d_in[8];  const float* b2b = (const float*)d_in[9];
  const float* w3a = (const float*)d_in[10]; const float* b3a = (const float*)d_in[11];
  const float* w3b = (const float*)d_in[12]; const float* b3b = (const float*)d_in[13];
  const float* fw  = (const float*)d_in[14]; const float* fb  = (const float*)d_in[15];
  float* out = (float*)d_out;

  const int N = N_NODES;
  const long long E = N_EDGES;

  // workspace carve-up (~231 MB total; 256B-aligned slices)
  char* ws = (char*)d_ws;
  size_t off = 0;
  auto take = [&](size_t bytes) -> char* {
    char* p = ws + off;
    off += (bytes + 255) & ~(size_t)255;
    return p;
  };
  _Float16* W1t  = (_Float16*)take((size_t)256 * 32 * 2);
  _Float16* W2t  = (_Float16*)take((size_t)256 * 128 * 2);
  _Float16* W3t  = (_Float16*)take((size_t)256 * 128 * 2);
  _Float16* Wb1t = (_Float16*)take((size_t)128 * 128 * 2);
  _Float16* Wb2t = (_Float16*)take((size_t)128 * 128 * 2);
  _Float16* Wb3t = (_Float16*)take((size_t)128 * 128 * 2);
  _Float16* Xh   = (_Float16*)take((size_t)N * 128 * 2);
  _Float16* PQ   = (_Float16*)take((size_t)N * 256 * 2);
  float*    XL   = (float*)take((size_t)3 * N * 128 * 4);  // x2|x4|x6 contiguous
  float* X2 = XL;
  float* X4 = XL + (size_t)N * 128;
  float* X6 = XL + (size_t)2 * N * 128;

  // 1) zero aggregation buffers (atomic-max targets; also gives 0 for edge-less nodes)
  {
    size_t n = (size_t)3 * N * 128;
    zero_f32<<<(int)((n + 255) / 256), 256, 0, stream>>>(XL, n);
  }
  // 2) weight prep (f32 -> transposed/padded f16)
  prep_wa<<<(256 * 32 + 255) / 256, 256, 0, stream>>>(w1a, W1t, 24, 32);
  prep_wa<<<(256 * 128 + 255) / 256, 256, 0, stream>>>(w2a, W2t, 128, 128);
  prep_wa<<<(256 * 128 + 255) / 256, 256, 0, stream>>>(w3a, W3t, 128, 128);
  prep_wb<<<(128 * 128 + 255) / 256, 256, 0, stream>>>(w1b, Wb1t);
  prep_wb<<<(128 * 128 + 255) / 256, 256, 0, stream>>>(w2b, Wb2t);
  prep_wb<<<(128 * 128 + 255) / 256, 256, 0, stream>>>(w3b, Wb3t);
  // 3) input features -> f16 padded [N,32]
  cvt_x_in<<<(N * 32 + 255) / 256, 256, 0, stream>>>(x, Xh);

  const int nodeBlocks = (N + 63) / 64;                 // 4 waves x 16 rows / block
  const int edgeBlocks = (int)((E + 63) / 64);          // 4 waves x 16 edges / block
  const size_t nfeat = (size_t)N * 128;

  // ---- layer 1 ----
  node_gemm<32><<<nodeBlocks, 128, 0, stream>>>(Xh, W1t, PQ, N);
  edge_conv<<<edgeBlocks, 128, 0, stream>>>(PQ, ei, b1a, Wb1t, b1b, X2, E);
  cvt_f32_f16<<<(int)((nfeat + 255) / 256), 256, 0, stream>>>(X2, Xh, nfeat);
  // ---- layer 2 ----
  node_gemm<128><<<nodeBlocks, 128, 0, stream>>>(Xh, W2t, PQ, N);
  edge_conv<<<edgeBlocks, 128, 0, stream>>>(PQ, ei, b2a, Wb2t, b2b, X4, E);
  cvt_f32_f16<<<(int)((nfeat + 255) / 256), 256, 0, stream>>>(X4, Xh, nfeat);
  // ---- layer 3 ----
  node_gemm<128><<<nodeBlocks, 128, 0, stream>>>(Xh, W3t, PQ, N);
  edge_conv<<<edgeBlocks, 128, 0, stream>>>(PQ, ei, b3a, Wb3t, b3b, X6, E);

  // ---- head ----
  final_head<<<(N + 255) / 256, 256, 0, stream>>>(X2, X4, X6, fw, fb, out, N);
}
